// GAT_60533269070023
// MI455X (gfx1250) — matrix operand order repaired and checked by reading, not compiled, not run on hardware
//
#include <hip/hip_runtime.h>
#include <hip/hip_bf16.h>

#define N_NODES 16384
#define NFEAT   256
#define NHID    32
#define NHEADS  4
#define NCLASS  32
#define HID     (NHEADS * NHID)   // 128

typedef __attribute__((ext_vector_type(16))) __bf16 v16bf;
typedef __attribute__((ext_vector_type(8)))  __bf16 v8bf;
typedef __attribute__((ext_vector_type(8)))  float  v8f;
typedef __attribute__((ext_vector_type(4)))  float  v4f;

__device__ __forceinline__ float elu_f(float x) {
  return x > 0.f ? x : (__expf(x) - 1.f);
}

// ---------------------------------------------------------------------------
// Feature transform: F_T[c][m] = sum_f Xin[m,f] * W[h,f,o]   (c = h*32+o)
// Output is COLUMN-major bf16 so the big GEMM can load WMMA B fragments as
// contiguous 16-byte chunks (K fastest within a column).
// ---------------------------------------------------------------------------
__global__ void __launch_bounds__(HID)
feat_kernel(const float* __restrict__ Xin,
            const float* __restrict__ W,
            __bf16* __restrict__ Ft,
            int nf)
{
  __shared__ float xs[NFEAT];
  const int m = blockIdx.x;
  const int c = threadIdx.x;                 // 0..127
  for (int f = c; f < nf; f += HID)
    xs[f] = Xin[(size_t)m * nf + f];
  __syncthreads();

  const int h = c >> 5, o = c & 31;
  const float* w = W + (size_t)h * nf * NHID + o;
  float acc = 0.f;
  #pragma unroll 8
  for (int f = 0; f < nf; ++f)
    acc = fmaf(xs[f], w[(size_t)f * NHID], acc);

  Ft[(size_t)c * N_NODES + m] = (__bf16)acc;
}

// ---------------------------------------------------------------------------
// Big GEMM helpers
// ---------------------------------------------------------------------------
__device__ __forceinline__ void load_a(const float* __restrict__ aptr, int kb,
                                       v4f (&a)[4])
{
  a[0] = *(const v4f*)(aptr + kb);
  a[1] = *(const v4f*)(aptr + kb + 4);
  a[2] = *(const v4f*)(aptr + kb + 16);
  a[3] = *(const v4f*)(aptr + kb + 20);
}

__device__ __forceinline__ void load_b(const __bf16* __restrict__ bptr, int kb,
                                       v8bf (&b)[8][2])
{
  #pragma unroll
  for (int t = 0; t < 8; ++t) {
    const __bf16* p = bptr + (size_t)t * 16 * N_NODES + kb;
    b[t][0] = *(const v8bf*)p;
    b[t][1] = *(const v8bf*)(p + 16);
  }
}

__device__ __forceinline__ void do_tiles(const v4f (&a)[4], const v8bf (&b)[8][2],
                                         v8f (&acc)[8])
{
  v16bf af;
  #pragma unroll
  for (int i = 0; i < 4; ++i) {
    af[i]      = (__bf16)a[0][i];
    af[4 + i]  = (__bf16)a[1][i];
    af[8 + i]  = (__bf16)a[2][i];
    af[12 + i] = (__bf16)a[3][i];
  }
  #pragma unroll
  for (int t = 0; t < 8; ++t) {
    v16bf bfrag;
    #pragma unroll
    for (int i = 0; i < 8; ++i) { bfrag[i] = b[t][0][i]; bfrag[8 + i] = b[t][1][i]; }
    acc[t] = __builtin_amdgcn_wmma_f32_16x16x32_bf16(
        false, af, false, bfrag, (short)0, acc[t], false, false);
  }
}

// ---------------------------------------------------------------------------
// Big GEMM: Xout[m, c] = elu( sum_k adj[m,k] * Ft[c][k] )
// adj fp32 row-major (streamed once from HBM, converted to bf16 in-register),
// Ft bf16 column-major (L2/WGP$-resident).
// One wave = 16 rows x 128 cols = 8 WMMA C tiles; software-pipelined K loop
// with ping-pong double buffers for A and B fragments. sched_barrier(0)
// fences keep the loads of step k+1 above the WMMAs of step k so the
// s_wait_loadcnt before each compute phase is satisfied a full phase early.
// ---------------------------------------------------------------------------
__global__ void __launch_bounds__(128)
adj_gemm_elu_kernel(const float* __restrict__ adj,
                    const __bf16* __restrict__ Ft,
                    float* __restrict__ Xout)
{
  const int lane = threadIdx.x & 31;
  const int wave = threadIdx.x >> 5;         // 0..3
  const int half = lane >> 4;                // 0 / 1
  const int ln   = lane & 15;
  const int rowBase = blockIdx.x * 64 + wave * 16;

  v8f acc[8];
  #pragma unroll
  for (int t = 0; t < 8; ++t)
    acc[t] = (v8f){0.f, 0.f, 0.f, 0.f, 0.f, 0.f, 0.f, 0.f};

  // A-fragment per ISA: lane<16 owns K {0..7,16..23}, lane>=16 owns {8..15,24..31}
  const float*  aptr = adj + (size_t)(rowBase + ln) * N_NODES + half * 8;
  // B-fragment: col = t*16+ln, same K split as A: {0..7,16..23} / {8..15,24..31}
  const __bf16* bptr = Ft + (size_t)ln * N_NODES + half * 8;

  v4f  a0[4], a1[4];
  v8bf b0[8][2], b1[8][2];

  // prime the pipeline with K-step 0
  load_a(aptr, 0, a0);
  load_b(bptr, 0, b0);

  for (int kb = 0; kb < N_NODES; kb += 64) {
    // prefetch the adj stream ahead of the double-buffer window
    const int kpf = (kb + 2048 < N_NODES) ? kb + 2048 : 0;
    __builtin_prefetch(aptr + kpf, 0, 1);

    // ---- issue loads for K-step kb+32 ----
    load_a(aptr, kb + 32, a1);
    load_b(bptr, kb + 32, b1);
    __builtin_amdgcn_sched_barrier(0);
    // ---- compute K-step kb (data loaded a full phase ago) ----
    do_tiles(a0, b0, acc);
    __builtin_amdgcn_sched_barrier(0);

    // ---- issue loads for K-step kb+64 (clamped at the tail) ----
    const int kn = (kb + 64 < N_NODES) ? kb + 64 : 0;
    load_a(aptr, kn, a0);
    load_b(bptr, kn, b0);
    __builtin_amdgcn_sched_barrier(0);
    // ---- compute K-step kb+32 ----
    do_tiles(a1, b1, acc);
    __builtin_amdgcn_sched_barrier(0);
  }

  // Epilogue: C layout VGPR v -> M = v + 8*half, N = ln ; fused ELU, row-major out
  #pragma unroll
  for (int t = 0; t < 8; ++t) {
    const int cc = t * 16 + ln;
    #pragma unroll
    for (int v = 0; v < 8; ++v) {
      const int rr = rowBase + v + half * 8;
      Xout[(size_t)rr * HID + cc] = elu_f(acc[t][v]);
    }
  }
}

// ---------------------------------------------------------------------------
// Output head: logits = elu(X @ Wout + b); out = log_softmax(logits)
// One wave32 per node row; shuffle reductions for max / sum-exp.
// ---------------------------------------------------------------------------
__global__ void __launch_bounds__(NCLASS)
head_kernel(const float* __restrict__ X,
            const float* __restrict__ Wout,
            const float* __restrict__ bout,
            float* __restrict__ out)
{
  const int m = blockIdx.x;
  const int j = threadIdx.x;                 // 0..31
  const float* xr = X + (size_t)m * HID;

  float s = bout[j];
  #pragma unroll 4
  for (int c = 0; c < HID; ++c)
    s = fmaf(xr[c], Wout[(size_t)c * NCLASS + j], s);

  float v = elu_f(s);

  float mx = v;
  #pragma unroll
  for (int o = 16; o >= 1; o >>= 1)
    mx = fmaxf(mx, __shfl_xor(mx, o, 32));

  float e  = __expf(v - mx);
  float se = e;
  #pragma unroll
  for (int o = 16; o >= 1; o >>= 1)
    se += __shfl_xor(se, o, 32);

  out[(size_t)m * NCLASS + j] = (v - mx) - __logf(se);
}

// ---------------------------------------------------------------------------
extern "C" void kernel_launch(void* const* d_in, const int* in_sizes, int n_in,
                              void* d_out, int out_size, void* d_ws, size_t ws_size,
                              hipStream_t stream)
{
  const float* x    = (const float*)d_in[0];   // [16384, 256]
  const float* adj  = (const float*)d_in[1];   // [16384, 16384]
  const float* W1   = (const float*)d_in[2];   // [4, 256, 32]
  const float* W2   = (const float*)d_in[3];   // [4, 128, 32]
  const float* Wout = (const float*)d_in[4];   // [128, 32]
  const float* bout = (const float*)d_in[5];   // [32]
  float* out = (float*)d_out;                  // [16384, 32]

  // workspace: Ft (bf16, col-major, 4 MiB) | X (fp32, row-major, 8 MiB)
  __bf16* Ft = (__bf16*)d_ws;
  float*  X  = (float*)((char*)d_ws + (size_t)HID * N_NODES * sizeof(__bf16));

  // layer 1
  feat_kernel<<<N_NODES, HID, 0, stream>>>(x, W1, Ft, NFEAT);
  adj_gemm_elu_kernel<<<N_NODES / 64, 128, 0, stream>>>(adj, Ft, X);
  // layer 2
  feat_kernel<<<N_NODES, HID, 0, stream>>>(X, W2, Ft, HID);
  adj_gemm_elu_kernel<<<N_NODES / 64, 128, 0, stream>>>(adj, Ft, X);
  // output head
  head_kernel<<<N_NODES, NCLASS, 0, stream>>>(X, Wout, bout, out);
}
